// ViT_IWM_60945585930442
// MI455X (gfx1250) — matrix operand order repaired and checked by reading, not compiled, not run on hardware
//
#include <hip/hip_runtime.h>
#include <math.h>

// ---------------- types ----------------
typedef __attribute__((ext_vector_type(16))) _Float16 v16h;
typedef __attribute__((ext_vector_type(8)))  float    v8f;
typedef __attribute__((ext_vector_type(4)))  _Float16 h4;

// ---------------- model dims ----------------
#define BB   64          // B = B_S * SF
#define NT   192         // tokens
#define ROWS (BB * NT)   // 12288
#define DE   768
#define DP   384
#define MME  48          // masked tokens
#define BS_  16          // pre-stack batch
#define OUT_T 12
#define KPAD 800         // mask-MLP first layer K (770 zero-padded to %32==0)

// ---------------- fragment loaders (contiguous k = 16*(lane/16) + s, same for A and B) ----------------
__device__ __forceinline__ v16h load_afrag(const _Float16* ap, int abase) {
    v16h f;
    #pragma unroll
    for (int j = 0; j < 4; ++j) {
        f[2 * j]         = ap[abase + 2 * j];
        f[2 * j + 1]     = ap[abase + 2 * j + 1];
        f[8 + 2 * j]     = ap[abase + 8 + 2 * j];
        f[8 + 2 * j + 1] = ap[abase + 8 + 2 * j + 1];
    }
    return f;
}
__device__ __forceinline__ v16h load_bfrag(const _Float16* bp, int bbase) {
    v16h f;
    #pragma unroll
    for (int e = 0; e < 16; ++e) f[e] = bp[bbase + e];
    return f;
}

// ---------------- GEMM: C = act(A @ W + bias) + resid ----------------
// A: MxK f32 row-major, W: KxN f32 row-major. WMMA f16 inputs, f32 accum.
// Contract: M % 128 == 0, N % 128 == 0, K % 32 == 0.
#define TM 128
#define TN 128
#define TK 32
#define LSTR 36   // LDS row stride in halfs (72B: 8B-aligned rows, conflict-free frag reads)

template<int ACT, bool HAS_BIAS, bool HAS_RES>
__global__ __launch_bounds__(256)
void gemm_wmma_f16(const float* __restrict__ A, const float* __restrict__ W,
                   const float* __restrict__ bias, const float* __restrict__ resid,
                   float* __restrict__ C, int M, int N, int K)
{
    __shared__ _Float16 As[TM * LSTR];   // [row][k]
    __shared__ _Float16 Bs[TN * LSTR];   // [n][k]  (transposed)

    const int tid  = threadIdx.x;
    const int lane = tid & 31;
    const int wave = tid >> 5;
    const int wm   = wave & 3;     // M sub-block (32 rows)
    const int wn   = wave >> 2;    // N sub-block (64 cols)
    const int blockM = blockIdx.y * TM;
    const int blockN = blockIdx.x * TN;

    const int r16   = lane & 15;
    const int hi    = lane >> 4;         // 0 / 1
    const int abase = hi ? 16 : 0;       // A fragment K base
    const int bbase = hi ? 16 : 0;       // B fragment K base

    const int a_kc   = (tid & 7) * 4;    // 0..28
    const int a_row0 = tid >> 3;         // 0..31
    const int b_n0   = (tid & 31) * 4;   // 0..124
    const int b_kr0  = tid >> 5;         // 0..7

    v8f acc[2][4];
    #pragma unroll
    for (int i = 0; i < 2; ++i)
        #pragma unroll
        for (int j = 0; j < 4; ++j)
            #pragma unroll
            for (int e = 0; e < 8; ++e) acc[i][j][e] = 0.0f;

    float4 aReg[4], bReg[4];

    auto load_global = [&](int k0) {
        #pragma unroll
        for (int pass = 0; pass < 4; ++pass) {
            const int row = a_row0 + pass * 32;
            aReg[pass] = *(const float4*)(A + (size_t)(blockM + row) * K + k0 + a_kc);
        }
        #pragma unroll
        for (int pass = 0; pass < 4; ++pass) {
            const int kr = b_kr0 + pass * 8;
            bReg[pass] = *(const float4*)(W + (size_t)(k0 + kr) * N + blockN + b_n0);
        }
    };
    auto store_lds = [&]() {
        #pragma unroll
        for (int pass = 0; pass < 4; ++pass) {
            const int row = a_row0 + pass * 32;
            h4 hv;
            hv[0] = (_Float16)aReg[pass].x; hv[1] = (_Float16)aReg[pass].y;
            hv[2] = (_Float16)aReg[pass].z; hv[3] = (_Float16)aReg[pass].w;
            *(h4*)&As[row * LSTR + a_kc] = hv;
        }
        #pragma unroll
        for (int pass = 0; pass < 4; ++pass) {
            const int kr = b_kr0 + pass * 8;
            Bs[(b_n0 + 0) * LSTR + kr] = (_Float16)bReg[pass].x;
            Bs[(b_n0 + 1) * LSTR + kr] = (_Float16)bReg[pass].y;
            Bs[(b_n0 + 2) * LSTR + kr] = (_Float16)bReg[pass].z;
            Bs[(b_n0 + 3) * LSTR + kr] = (_Float16)bReg[pass].w;
        }
    };

    load_global(0);
    store_lds();

    for (int k0 = 0; k0 < K; k0 += TK) {
        __syncthreads();

        v16h aF[2], bF[4];
        #pragma unroll
        for (int i = 0; i < 2; ++i)
            aF[i] = load_afrag(&As[(wm * 32 + i * 16 + r16) * LSTR], abase);
        #pragma unroll
        for (int j = 0; j < 4; ++j)
            bF[j] = load_bfrag(&Bs[(wn * 64 + j * 16 + r16) * LSTR], bbase);

        const bool more = (k0 + TK) < K;
        if (more) load_global(k0 + TK);

        #pragma unroll
        for (int i = 0; i < 2; ++i)
            #pragma unroll
            for (int j = 0; j < 4; ++j)
                acc[i][j] = __builtin_amdgcn_wmma_f32_16x16x32_f16(
                    false, aF[i], false, bF[j], (short)0, acc[i][j], false, false);

        __syncthreads();
        if (more) store_lds();
    }

    #pragma unroll
    for (int i = 0; i < 2; ++i) {
        #pragma unroll
        for (int j = 0; j < 4; ++j) {
            const int col = blockN + wn * 64 + j * 16 + r16;
            float bv = 0.0f;
            if (HAS_BIAS) bv = bias[col];
            #pragma unroll
            for (int vi = 0; vi < 8; ++vi) {
                const int row = blockM + wm * 32 + i * 16 + hi * 8 + vi;
                const size_t off = (size_t)row * N + col;
                float v = acc[i][j][vi];
                if (HAS_BIAS) v += bv;
                if (ACT == 1) {
                    v = fmaxf(v, 0.0f);
                } else if (ACT == 2) {
                    const float t = v;
                    const float z = 0.7978845608f * (t + 0.044715f * t * t * t);
                    v = t / (1.0f + __expf(-2.0f * z));
                }
                if (HAS_RES) v += resid[off];
                C[off] = v;
            }
        }
    }
}

// ---------------- WMMA flash attention ----------------
// One block per (b, h, 64-query chunk). dh = 64, keys streamed in 3 chunks of 64.
#define ASTR 72   // half stride for 64-wide f16 tiles
#define SSTR 66   // float stride for the score tile

__global__ __launch_bounds__(256)
void attn_wmma_k(const float* __restrict__ qkv, float* __restrict__ out,
                 int nh, int stride)
{
    __shared__ _Float16 Qs[64 * ASTR];
    __shared__ _Float16 Ks[64 * ASTR];
    __shared__ _Float16 Vs[64 * ASTR];   // [d][key] (transposed)
    __shared__ _Float16 Ps[64 * ASTR];
    __shared__ float    Sf[64 * SSTR];
    __shared__ float    pmax[64 * 4];
    __shared__ float    psum[64 * 4];
    __shared__ float    Mrow[64], Srow[64], Frow[64];

    const int blk = blockIdx.x;
    const int qc  = blk % 3;
    const int h   = (blk / 3) % nh;
    const int b   = blk / (3 * nh);

    const int tid  = threadIdx.x;
    const int lane = tid & 31;
    const int wave = tid >> 5;
    const int wm   = wave & 3;       // 16 q-rows
    const int wn   = wave >> 2;      // 32-col half
    const int r16  = lane & 15;
    const int hi   = lane >> 4;
    const int abase = hi ? 16 : 0;
    const int bbase = hi ? 16 : 0;

    const int qoff = h * 64;
    const int koff = nh * 64 + h * 64;
    const int voff = 2 * nh * 64 + h * 64;
    const size_t rowbase = (size_t)b * NT * stride;

    const int d4 = (tid & 15) * 4;   // staging: 4 dh-cols
    const int sr0 = tid >> 4;        // staging: row 0..15 (x4 passes)

    // stage Q chunk (64 x 64): batch loads, then stores
    {
        float4 qReg[4];
        #pragma unroll
        for (int pass = 0; pass < 4; ++pass) {
            const int r = sr0 + pass * 16;
            qReg[pass] = *(const float4*)(qkv + rowbase + (size_t)(qc * 64 + r) * stride + qoff + d4);
        }
        #pragma unroll
        for (int pass = 0; pass < 4; ++pass) {
            const int r = sr0 + pass * 16;
            h4 hv; hv[0] = (_Float16)qReg[pass].x; hv[1] = (_Float16)qReg[pass].y;
            hv[2] = (_Float16)qReg[pass].z;        hv[3] = (_Float16)qReg[pass].w;
            *(h4*)&Qs[r * ASTR + d4] = hv;
        }
    }
    if (tid < 64) { Mrow[tid] = -3.0e38f; Srow[tid] = 0.0f; }

    v8f accO[2];
    #pragma unroll
    for (int jt = 0; jt < 2; ++jt)
        #pragma unroll
        for (int e = 0; e < 8; ++e) accO[jt][e] = 0.0f;

    for (int kc = 0; kc < 3; ++kc) {
        __syncthreads();   // prev chunk's WMMAs done before restaging K/V
        // stage K chunk [key][d] and V chunk transposed [d][key]: batch loads, then stores
        {
            float4 kReg[4], vReg[4];
            #pragma unroll
            for (int pass = 0; pass < 4; ++pass) {
                const int r = sr0 + pass * 16;       // key within chunk
                const float* src = qkv + rowbase + (size_t)(kc * 64 + r) * stride;
                kReg[pass] = *(const float4*)(src + koff + d4);
                vReg[pass] = *(const float4*)(src + voff + d4);
            }
            #pragma unroll
            for (int pass = 0; pass < 4; ++pass) {
                const int r = sr0 + pass * 16;
                h4 hv; hv[0] = (_Float16)kReg[pass].x; hv[1] = (_Float16)kReg[pass].y;
                hv[2] = (_Float16)kReg[pass].z;        hv[3] = (_Float16)kReg[pass].w;
                *(h4*)&Ks[r * ASTR + d4] = hv;
                Vs[(d4 + 0) * ASTR + r] = (_Float16)vReg[pass].x;
                Vs[(d4 + 1) * ASTR + r] = (_Float16)vReg[pass].y;
                Vs[(d4 + 2) * ASTR + r] = (_Float16)vReg[pass].z;
                Vs[(d4 + 3) * ASTR + r] = (_Float16)vReg[pass].w;
            }
        }
        __syncthreads();

        // S = Q @ Kc^T  (wave: 16 rows x 32 keys); preload all fragments, then 4 WMMAs
        v8f accS[2];
        #pragma unroll
        for (int jt = 0; jt < 2; ++jt)
            #pragma unroll
            for (int e = 0; e < 8; ++e) accS[jt][e] = 0.0f;
        {
            v16h aF[2], bF[2][2];
            #pragma unroll
            for (int kk = 0; kk < 2; ++kk) {
                aF[kk] = load_afrag(&Qs[(wm * 16 + r16) * ASTR + kk * 32], abase);
                #pragma unroll
                for (int jt = 0; jt < 2; ++jt)
                    bF[kk][jt] = load_bfrag(&Ks[(wn * 32 + jt * 16 + r16) * ASTR + kk * 32], bbase);
            }
            #pragma unroll
            for (int kk = 0; kk < 2; ++kk)
                #pragma unroll
                for (int jt = 0; jt < 2; ++jt)
                    accS[jt] = __builtin_amdgcn_wmma_f32_16x16x32_f16(
                        false, aF[kk], false, bF[kk][jt], (short)0, accS[jt], false, false);
        }
        #pragma unroll
        for (int jt = 0; jt < 2; ++jt)
            #pragma unroll
            for (int vi = 0; vi < 8; ++vi)
                Sf[(wm * 16 + hi * 8 + vi) * SSTR + wn * 32 + jt * 16 + r16] = accS[jt][vi] * 0.125f;
        __syncthreads();

        // online softmax update: 4 threads per row
        {
            const int row  = tid >> 2;
            const int part = tid & 3;
            const float* srow = &Sf[row * SSTR + part * 16];
            float lmax = -3.0e38f;
            #pragma unroll
            for (int c = 0; c < 16; ++c) lmax = fmaxf(lmax, srow[c]);
            pmax[row * 4 + part] = lmax;
            __syncthreads();
            const float cmax = fmaxf(fmaxf(pmax[row * 4 + 0], pmax[row * 4 + 1]),
                                     fmaxf(pmax[row * 4 + 2], pmax[row * 4 + 3]));
            const float mold = Mrow[row];
            const float mnew = fmaxf(mold, cmax);
            float lsum = 0.0f;
            #pragma unroll
            for (int c = 0; c < 16; ++c) {
                const float p = __expf(srow[c] - mnew);
                Ps[row * ASTR + part * 16 + c] = (_Float16)p;
                lsum += p;
            }
            psum[row * 4 + part] = lsum;
            __syncthreads();
            if (part == 0) {
                const float csum = psum[row * 4 + 0] + psum[row * 4 + 1]
                                 + psum[row * 4 + 2] + psum[row * 4 + 3];
                const float f = __expf(mold - mnew);
                Srow[row] = Srow[row] * f + csum;
                Mrow[row] = mnew;
                Frow[row] = f;
            }
        }
        __syncthreads();

        // O = O*f + P @ Vc  (wave: 16 rows x 32 dh-cols); preload fragments, then 4 WMMAs
        #pragma unroll
        for (int vi = 0; vi < 8; ++vi) {
            const float f = Frow[wm * 16 + hi * 8 + vi];
            accO[0][vi] *= f;
            accO[1][vi] *= f;
        }
        {
            v16h aF[2], bF[2][2];
            #pragma unroll
            for (int kk = 0; kk < 2; ++kk) {
                aF[kk] = load_afrag(&Ps[(wm * 16 + r16) * ASTR + kk * 32], abase);
                #pragma unroll
                for (int jt = 0; jt < 2; ++jt)
                    bF[kk][jt] = load_bfrag(&Vs[(wn * 32 + jt * 16 + r16) * ASTR + kk * 32], bbase);
            }
            #pragma unroll
            for (int kk = 0; kk < 2; ++kk)
                #pragma unroll
                for (int jt = 0; jt < 2; ++jt)
                    accO[jt] = __builtin_amdgcn_wmma_f32_16x16x32_f16(
                        false, aF[kk], false, bF[kk][jt], (short)0, accO[jt], false, false);
        }
    }

    // write out: divide by running sum
    const int D = nh * 64;
    #pragma unroll
    for (int vi = 0; vi < 8; ++vi) {
        const int row = wm * 16 + hi * 8 + vi;
        const float inv = 1.0f / Srow[row];
        float* orow = out + (size_t)(b * NT + qc * 64 + row) * D + h * 64 + wn * 32;
        orow[r16]      = accO[0][vi] * inv;
        orow[16 + r16] = accO[1][vi] * inv;
    }
}

// ---------------- LayerNorm (one block per row) ----------------
__global__ __launch_bounds__(256)
void layernorm_k(const float* __restrict__ x, const float* __restrict__ g,
                 const float* __restrict__ b, float* __restrict__ out, int D)
{
    __shared__ float s1[256], s2[256];
    const int row = blockIdx.x;
    const float* xr = x + (size_t)row * D;
    float sum = 0.0f, sq = 0.0f;
    for (int i = threadIdx.x; i < D; i += 256) {
        const float v = xr[i];
        sum += v; sq += v * v;
    }
    s1[threadIdx.x] = sum; s2[threadIdx.x] = sq;
    __syncthreads();
    for (int s = 128; s > 0; s >>= 1) {
        if (threadIdx.x < s) { s1[threadIdx.x] += s1[threadIdx.x + s]; s2[threadIdx.x] += s2[threadIdx.x + s]; }
        __syncthreads();
    }
    const float mean = s1[0] / (float)D;
    const float var  = s2[0] / (float)D - mean * mean;
    const float rinv = rsqrtf(var + 1e-5f);
    float* orow = out + (size_t)row * D;
    for (int i = threadIdx.x; i < D; i += 256)
        orow[i] = (xr[i] - mean) * rinv * g[i] + b[i];
}

// ---------------- add positional embedding ----------------
__global__ void add_pos_k(float* __restrict__ x, const float* __restrict__ pos, int D)
{
    const size_t i = (size_t)blockIdx.x * 256 + threadIdx.x;
    const size_t total = (size_t)ROWS * D;
    if (i < total) x[i] += pos[i % ((size_t)NT * D)];
}

// ---------------- mask-token MLP input: relu(concat(mask_vec, aug)), K-padded ----------------
__global__ __launch_bounds__(256)
void build_mask_in_k(const float* __restrict__ mask_W, const float* __restrict__ mask_b,
                     const float* __restrict__ mask_emb, const float* __restrict__ aug,
                     const int* __restrict__ mask_pos, float* __restrict__ out)
{
    const int r = blockIdx.x;              // 0..3071
    const int b = r / MME, m = r % MME;
    const int bs = b / 4;                  // SF=4
    const int pos = mask_pos[bs * MME + m];
    float* o = out + (size_t)r * KPAD;
    for (int d = threadIdx.x; d < DE; d += 256) {
        const float v = mask_W[d] + mask_b[d] + mask_emb[pos * DE + d];
        o[d] = fmaxf(v, 0.0f);
    }
    for (int d = DE + threadIdx.x; d < KPAD; d += 256) {
        float v = 0.0f;
        if (d < DE + 2) v = fmaxf(aug[bs * 2 + (d - DE)], 0.0f);
        o[d] = v;
    }
}

// ---------------- zero-pad mm_W0 (770 x 768) -> (800 x 768) ----------------
__global__ void pad_w0_k(const float* __restrict__ w0, float* __restrict__ out)
{
    const int i = blockIdx.x * 256 + threadIdx.x;
    const int total = KPAD * DE;
    if (i < total) {
        const int row = i / DE;
        out[i] = (row < DE + 2) ? w0[i] : 0.0f;
    }
}

// ---------------- scatter mask rows into Xp ----------------
__global__ __launch_bounds__(256)
void scatter_mask_k(float* __restrict__ Xp, const float* __restrict__ h,
                    const int* __restrict__ mask_pos)
{
    const int r = blockIdx.x;              // 0..3071
    const int b = r / MME, m = r % MME;
    const int bs = b / 4;
    const int idx = mask_pos[bs * MME + m] + bs * BS_;   // faithful "* B_S" arithmetic
    float* dst = Xp + (size_t)idx * DE;
    const float* src = h + (size_t)r * DE;
    for (int d = threadIdx.x; d < DE; d += 256) dst[d] = src[d];
}

// ---------------- final gather of Xo and y ----------------
__global__ __launch_bounds__(256)
void gather_k(const float* __restrict__ Xo, const float* __restrict__ y,
              const int* __restrict__ mask_pos, float* __restrict__ out)
{
    const int r = blockIdx.x;              // 0..767  (B * OUT_T)
    const int b = r / OUT_T, t = r % OUT_T;
    const int pos = mask_pos[(b >> 2) * MME + (b & 3) * OUT_T + t];
    const float* xsrc = Xo + ((size_t)(b * NT + pos)) * DP;
    const float* ysrc = y  + ((size_t)(b * NT + pos)) * DP;
    float* o0 = out + (size_t)r * DP;
    float* o1 = out + (size_t)(BB * OUT_T + r) * DP;
    for (int d = threadIdx.x; d < DP; d += 256) { o0[d] = xsrc[d]; o1[d] = ysrc[d]; }
}

// ---------------- host orchestration ----------------
extern "C" void kernel_launch(void* const* d_in, const int* in_sizes, int n_in,
                              void* d_out, int out_size, void* d_ws, size_t ws_size,
                              hipStream_t stream)
{
    const float* X         = (const float*)d_in[0];
    const float* y         = (const float*)d_in[1];
    const float* aug       = (const float*)d_in[2];
    const int*   mask_pos  = (const int*)d_in[3];
    const float* proj_W    = (const float*)d_in[4];
    const float* enc_pos   = (const float*)d_in[5];
    const float* enc_ln1_g = (const float*)d_in[6];
    const float* enc_ln1_b = (const float*)d_in[7];
    const float* enc_Wqkv  = (const float*)d_in[8];
    const float* enc_Wo    = (const float*)d_in[9];
    const float* enc_ln2_g = (const float*)d_in[10];
    const float* enc_ln2_b = (const float*)d_in[11];
    const float* enc_W1    = (const float*)d_in[12];
    const float* enc_W2    = (const float*)d_in[13];
    const float* enc_lnf_g = (const float*)d_in[14];
    const float* enc_lnf_b = (const float*)d_in[15];
    const float* mask_W    = (const float*)d_in[16];
    const float* mask_b    = (const float*)d_in[17];
    const float* mask_emb  = (const float*)d_in[18];
    const float* mm_W0     = (const float*)d_in[19];
    const float* mm_b0     = (const float*)d_in[20];
    const float* mm_W      = (const float*)d_in[21];
    const float* mm_b      = (const float*)d_in[22];
    const float* pp_W      = (const float*)d_in[23];
    const float* pp_b      = (const float*)d_in[24];
    const float* pred_pos  = (const float*)d_in[25];
    const float* pred_ln1_g= (const float*)d_in[26];
    const float* pred_ln1_b= (const float*)d_in[27];
    const float* pred_Wqkv = (const float*)d_in[28];
    const float* pred_Wo   = (const float*)d_in[29];
    const float* pred_ln2_g= (const float*)d_in[30];
    const float* pred_ln2_b= (const float*)d_in[31];
    const float* pred_W1   = (const float*)d_in[32];
    const float* pred_W2   = (const float*)d_in[33];
    const float* pred_lnf_g= (const float*)d_in[34];
    const float* pred_lnf_b= (const float*)d_in[35];

    // workspace layout (floats)
    float* bufA   = (float*)d_ws;                       // ROWS x DE (act x)
    float* bufH   = bufA   + (size_t)ROWS * DE;         // ROWS x DE (ln / attn out)
    float* bufBig = bufH   + (size_t)ROWS * DE;         // ROWS x 3072 (qkv / ffn mid)
    float* bufM0  = bufBig + (size_t)ROWS * 3072;       // 3072 x KPAD
    float* bufM1  = bufM0  + (size_t)3072 * KPAD;       // 3072 x 768
    float* bufM2  = bufM1  + (size_t)3072 * DE;         // 3072 x 768
    float* bufW0  = bufM2  + (size_t)3072 * DE;         // KPAD x 768 (padded mm_W0)

    auto gemm = [&](const float* A, const float* W, const float* bias,
                    const float* resid, float* C, int M, int N, int K, int act) {
        dim3 grid(N / TN, M / TM);
        if (resid)
            gemm_wmma_f16<0, false, true ><<<grid, 256, 0, stream>>>(A, W, nullptr, resid, C, M, N, K);
        else if (act == 2)
            gemm_wmma_f16<2, false, false><<<grid, 256, 0, stream>>>(A, W, nullptr, nullptr, C, M, N, K);
        else if (act == 1)
            gemm_wmma_f16<1, true,  false><<<grid, 256, 0, stream>>>(A, W, bias, nullptr, C, M, N, K);
        else if (bias)
            gemm_wmma_f16<0, true,  false><<<grid, 256, 0, stream>>>(A, W, bias, nullptr, C, M, N, K);
        else
            gemm_wmma_f16<0, false, false><<<grid, 256, 0, stream>>>(A, W, nullptr, nullptr, C, M, N, K);
    };
    auto ln = [&](const float* x, const float* g, const float* b, float* o, int D) {
        layernorm_k<<<ROWS, 256, 0, stream>>>(x, g, b, o, D);
    };

    // 1) Xp = X @ proj_W
    gemm(X, proj_W, nullptr, nullptr, bufA, ROWS, DE, DE, 0);

    // 2) mask-token MLP (first layer K zero-padded 770 -> 800)
    pad_w0_k<<<(KPAD * DE + 255) / 256, 256, 0, stream>>>(mm_W0, bufW0);
    build_mask_in_k<<<3072, 256, 0, stream>>>(mask_W, mask_b, mask_emb, aug, mask_pos, bufM0);
    gemm(bufM0, bufW0, mm_b0, nullptr, bufM1, 3072, DE, KPAD, 1);
    gemm(bufM1, mm_W + 0 * DE * DE, mm_b + 0 * DE, nullptr, bufM2, 3072, DE, DE, 1);
    gemm(bufM2, mm_W + 1 * DE * DE, mm_b + 1 * DE, nullptr, bufM1, 3072, DE, DE, 1);
    gemm(bufM1, mm_W + 2 * DE * DE, mm_b + 2 * DE, nullptr, bufM2, 3072, DE, DE, 1);

    // 3) scatter masked rows, add encoder pos
    scatter_mask_k<<<3072, 256, 0, stream>>>(bufA, bufM2, mask_pos);
    {
        const size_t total = (size_t)ROWS * DE;
        add_pos_k<<<(unsigned)((total + 255) / 256), 256, 0, stream>>>(bufA, enc_pos, DE);
    }

    // 4) encoder blocks (D=768, nh=12, FFN=3072)
    for (int l = 0; l < 6; ++l) {
        ln(bufA, enc_ln1_g + l * DE, enc_ln1_b + l * DE, bufH, DE);
        gemm(bufH, enc_Wqkv + (size_t)l * DE * 3 * DE, nullptr, nullptr, bufBig, ROWS, 3 * DE, DE, 0);
        attn_wmma_k<<<BB * 12 * 3, 256, 0, stream>>>(bufBig, bufH, 12, 3 * DE);
        gemm(bufH, enc_Wo + (size_t)l * DE * DE, nullptr, bufA, bufA, ROWS, DE, DE, 0);
        ln(bufA, enc_ln2_g + l * DE, enc_ln2_b + l * DE, bufH, DE);
        gemm(bufH, enc_W1 + (size_t)l * DE * 4 * DE, nullptr, nullptr, bufBig, ROWS, 4 * DE, DE, 2);
        gemm(bufBig, enc_W2 + (size_t)l * 4 * DE * DE, nullptr, bufA, bufA, ROWS, DE, 4 * DE, 0);
    }

    // 5) encoder final LN + predictor projection
    ln(bufA, enc_lnf_g, enc_lnf_b, bufH, DE);
    gemm(bufH, pp_W, pp_b, nullptr, bufA, ROWS, DP, DE, 0);
    {
        const size_t total = (size_t)ROWS * DP;
        add_pos_k<<<(unsigned)((total + 255) / 256), 256, 0, stream>>>(bufA, pred_pos, DP);
    }

    // 6) predictor blocks (D=384, nh=6, FFN=1536)
    for (int l = 0; l < 4; ++l) {
        ln(bufA, pred_ln1_g + l * DP, pred_ln1_b + l * DP, bufH, DP);
        gemm(bufH, pred_Wqkv + (size_t)l * DP * 3 * DP, nullptr, nullptr, bufBig, ROWS, 3 * DP, DP, 0);
        attn_wmma_k<<<BB * 6 * 3, 256, 0, stream>>>(bufBig, bufH, 6, 3 * DP);
        gemm(bufH, pred_Wo + (size_t)l * DP * DP, nullptr, bufA, bufA, ROWS, DP, DP, 0);
        ln(bufA, pred_ln2_g + l * DP, pred_ln2_b + l * DP, bufH, DP);
        gemm(bufH, pred_W1 + (size_t)l * DP * 4 * DP, nullptr, nullptr, bufBig, ROWS, 4 * DP, DP, 2);
        gemm(bufBig, pred_W2 + (size_t)l * 4 * DP * DP, nullptr, bufA, bufA, ROWS, DP, 4 * DP, 0);
    }

    // 7) predictor final LN + gathers
    ln(bufA, pred_lnf_g, pred_lnf_b, bufH, DP);
    gather_k<<<BB * OUT_T, 256, 0, stream>>>(bufH, y, mask_pos, (float*)d_out);
}
